// DyadicWinograd2D_7249904795702
// MI455X (gfx1250) — matrix operand order repaired and checked by reading, not compiled, not run on hardware
//
#include <hip/hip_runtime.h>
#include <math.h>

typedef unsigned short u16;
typedef __attribute__((ext_vector_type(16))) __bf16 v16bf;
typedef __attribute__((ext_vector_type(8)))  float  v8f;
typedef int dw_v2i __attribute__((vector_size(8)));   // b64 payload type

#define NN   10          // transform size N
#define MT   8           // M_TILE
#define THT  28
#define TWT  28
#define LT   (THT*TWT)   // 784 tiles per image
#define BB   8           // batch
#define MR   (BB*LT)     // 6272 GEMM rows
#define CC   64          // in channels  (K)
#define OO   64          // out channels (N)
#define FB   (NN*NN)     // 100 frequency bins
#define HWD  224
#define MBLK 128
#define NMB  (MR/MBLK)   // 49
#define XSTR 72          // padded LDS row stride (u16): 144B = 36 dwords.
                         // 16B-aligned rows (keeps ds_load_b128) and 36 mod 64
                         // spreads the 16 fragment rows across distinct banks.

// CDNA5 async LDS staging path (ASYNCcnt), guarded so either toolchain compiles.
#if defined(__gfx1250__) && \
    __has_builtin(__builtin_amdgcn_global_load_async_to_lds_b64) && \
    __has_builtin(__builtin_amdgcn_s_wait_asynccnt)
#define DW_ASYNC 1
// Builtin signature (from hipcc diagnostic): arg0 = AS1 v2i* (global src),
// arg1 = LDS v2i* (dst), then two immediate ints (offset, cpol).
#define TO_GLB(p) ((__attribute__((address_space(1))) dw_v2i*)(uintptr_t)(p))
#define TO_LDS(p) ((__attribute__((address_space(3))) dw_v2i*)(uintptr_t)(p))
#else
#define DW_ASYNC 0
#endif

__device__ __forceinline__ u16 f2bf(float f) {
  unsigned int u = __builtin_bit_cast(unsigned int, f);
  u += 0x7FFFu + ((u >> 16) & 1u);     // round-to-nearest-even
  return (u16)(u >> 16);
}
__device__ __forceinline__ __bf16 bfbits(u16 u) {
  return __builtin_bit_cast(__bf16, u);
}

// Fold the dyadic-quantized 3-shear lifting rotation into a 2x2 matrix,
// matching the numpy reference (angle reduction + negate flag + 2^-20 grid).
__device__ void dyadic_mats(int k, int i, bool inverse,
                            float& mrr, float& mri, float& mir, float& mii) {
  const double PI = 3.14159265358979323846;
  double sign = inverse ? 1.0 : -1.0;
  double ang = sign * 2.0 * PI * (double)(k * i) / (double)NN;
  double a = fmod(ang + PI, 2.0 * PI);
  if (a < 0.0) a += 2.0 * PI;
  a -= PI;
  bool neg = false;
  if (a >  PI * 0.5) { a -= PI; neg = true; }
  if (a < -PI * 0.5) { a += PI; neg = true; }
  const double S = 1048576.0;          // 2^20
  double lam = rint(-tan(a * 0.5) * S) / S;
  double gam = rint( sin(a)       * S) / S;
  double s = neg ? -1.0 : 1.0;
  mrr = (float)(s * (1.0 + lam * gam));
  mri = (float)(s * (2.0 * lam + gam * lam * lam));
  mir = (float)(s * gam);
  mii = (float)(s * (1.0 + gam * lam));
}

// ---------------- kernel 1: weights fp32 -> bf16, layout [f][c][o] ----------
__global__ __launch_bounds__(256) void dw_prep_weights(
    const float* __restrict__ wre_in, const float* __restrict__ wim_in,
    u16* __restrict__ wre, u16* __restrict__ wim, u16* __restrict__ wimn)
{
  int idx = blockIdx.x * 256 + threadIdx.x;
  if (idx >= FB * CC * OO) return;
  int o   = idx / (CC * FB);
  int rem = idx - o * (CC * FB);
  int c   = rem / FB;
  int f   = rem - c * FB;
  float re = wre_in[idx];
  float im = wim_in[idx];
  int w = (f * CC + c) * OO + o;
  wre[w]  = f2bf(re);
  wim[w]  = f2bf(im);
  wimn[w] = f2bf(-im);   // pre-negated: bf16 WMMA has no A/B negate (NEG=C only)
}

// ---------------- kernel 2: forward 2D transform -> bf16 X [f][m][c] --------
__global__ __launch_bounds__(256) void dw_fwd(
    const float* __restrict__ x, u16* __restrict__ Xre, u16* __restrict__ Xim)
{
  __shared__ float sMrr[FB], sMri[FB], sMir[FB], sMii[FB];
  __shared__ float sRe[CC * FB];   // 25.6 KB
  __shared__ float sIm[CC * FB];   // 25.6 KB
  int tid = threadIdx.x;
  if (tid < FB) {
    float a, b2, c2, d;
    dyadic_mats(tid / NN, tid % NN, false, a, b2, c2, d);
    sMrr[tid] = a; sMri[tid] = b2; sMir[tid] = c2; sMii[tid] = d;
  }
  __syncthreads();

  int m  = blockIdx.x;
  int b  = m / LT;
  int l  = m - b * LT;
  int ty = l / TWT;
  int tx = l - ty * TWT;
  const float* xb = x + (size_t)b * CC * HWD * HWD;

  // pass 1: row transform (along W) for each (channel, tile-row)
  for (int t = tid; t < CC * NN; t += 256) {
    int c = t / NN;
    int i = t - c * NN;
    int gy = ty * MT - 1 + i;
    bool rowok = (gy >= 0) && (gy < HWD);
    const float* row = xb + ((size_t)c * HWD + (gy < 0 ? 0 : (gy >= HWD ? HWD - 1 : gy))) * HWD;
    float v[NN];
    #pragma unroll
    for (int j = 0; j < NN; ++j) {
      int gx = tx * MT - 1 + j;
      v[j] = (rowok && gx >= 0 && gx < HWD) ? row[gx] : 0.0f;
    }
    #pragma unroll
    for (int k = 0; k < NN; ++k) {
      float re = 0.f, im = 0.f;
      #pragma unroll
      for (int j = 0; j < NN; ++j) {
        re += sMrr[k * NN + j] * v[j];
        im += sMir[k * NN + j] * v[j];
      }
      sRe[c * FB + i * NN + k] = re;
      sIm[c * FB + i * NN + k] = im;
    }
  }
  __syncthreads();

  // pass 2: column transform (along H), complex in, write bf16 X
  for (int t = tid; t < CC * NN; t += 256) {
    int c = t & (CC - 1);           // c fastest -> coalesced global stores
    int k = t >> 6;
    float re1[NN], im1[NN];
    #pragma unroll
    for (int i = 0; i < NN; ++i) {
      re1[i] = sRe[c * FB + i * NN + k];
      im1[i] = sIm[c * FB + i * NN + k];
    }
    #pragma unroll
    for (int r = 0; r < NN; ++r) {
      float re = 0.f, im = 0.f;
      #pragma unroll
      for (int i = 0; i < NN; ++i) {
        float mrr = sMrr[r * NN + i], mri = sMri[r * NN + i];
        float mir = sMir[r * NN + i], mii = sMii[r * NN + i];
        re += mrr * re1[i] + mri * im1[i];
        im += mir * re1[i] + mii * im1[i];
      }
      size_t xi = ((size_t)(r * NN + k) * MR + m) * CC + c;
      Xre[xi] = f2bf(re);
      Xim[xi] = f2bf(im);
    }
  }
}

// ---------------- kernel 3: per-bin complex GEMM via bf16 WMMA --------------
// Y[f] (MR x OO, f32) = X[f] (MR x CC, bf16-complex) * W[f] (CC x OO, bf16-complex)
__global__ __launch_bounds__(256) void dw_gemm(
    const u16* __restrict__ Xre, const u16* __restrict__ Xim,
    const u16* __restrict__ Wre, const u16* __restrict__ Wim, const u16* __restrict__ WimN,
    float* __restrict__ Yre, float* __restrict__ Yim)
{
  __shared__ __align__(16) u16 sXre[MBLK * XSTR];  // 18.0 KB
  __shared__ __align__(16) u16 sXim[MBLK * XSTR];  // 18.0 KB
  __shared__ __align__(16) u16 sWre[CC * XSTR];    // 9.0 KB
  __shared__ __align__(16) u16 sWim[CC * XSTR];    // 9.0 KB
  __shared__ __align__(16) u16 sWin[CC * XSTR];    // 9.0 KB  (total 63.0 KB)

  int f   = blockIdx.y;
  int mb  = blockIdx.x;
  int tid = threadIdx.x;

  size_t xoff = ((size_t)f * MR + (size_t)mb * MBLK) * CC;
  size_t woff = (size_t)f * CC * OO;

#if DW_ASYNC
  // CDNA5 async copy: global -> LDS without touching VGPRs (ASYNCcnt tracked).
  // Padded dst rows: 18 uint2 (72 u16) per 16-uint2 source row.
  const uint2* gXre = (const uint2*)(Xre  + xoff);
  const uint2* gXim = (const uint2*)(Xim  + xoff);
  const uint2* gWre = (const uint2*)(Wre  + woff);
  const uint2* gWim = (const uint2*)(Wim  + woff);
  const uint2* gWin = (const uint2*)(WimN + woff);
  for (int i = tid; i < MBLK * 16; i += 256) {
    int row = i >> 4, q = i & 15;
    int dst = row * 18 + q;
    int src = row * 16 + q;
    __builtin_amdgcn_global_load_async_to_lds_b64(
        TO_GLB(gXre + src), TO_LDS((uint2*)sXre + dst), 0, 0);
    __builtin_amdgcn_global_load_async_to_lds_b64(
        TO_GLB(gXim + src), TO_LDS((uint2*)sXim + dst), 0, 0);
  }
  for (int i = tid; i < CC * 16; i += 256) {
    int row = i >> 4, q = i & 15;
    int dst = row * 18 + q;
    int src = row * 16 + q;
    __builtin_amdgcn_global_load_async_to_lds_b64(
        TO_GLB(gWre + src), TO_LDS((uint2*)sWre + dst), 0, 0);
    __builtin_amdgcn_global_load_async_to_lds_b64(
        TO_GLB(gWim + src), TO_LDS((uint2*)sWim + dst), 0, 0);
    __builtin_amdgcn_global_load_async_to_lds_b64(
        TO_GLB(gWin + src), TO_LDS((uint2*)sWin + dst), 0, 0);
  }
  __builtin_amdgcn_s_wait_asynccnt(0);
#else
  // Sync fallback: uint4 copies, 9 uint4 per padded 72-u16 row (src has 8).
  const uint4* gXre = (const uint4*)(Xre  + xoff);
  const uint4* gXim = (const uint4*)(Xim  + xoff);
  const uint4* gWre = (const uint4*)(Wre  + woff);
  const uint4* gWim = (const uint4*)(Wim  + woff);
  const uint4* gWin = (const uint4*)(WimN + woff);
  for (int i = tid; i < MBLK * 8; i += 256) {
    int row = i >> 3, q = i & 7;
    ((uint4*)sXre)[row * 9 + q] = gXre[row * 8 + q];
    ((uint4*)sXim)[row * 9 + q] = gXim[row * 8 + q];
  }
  for (int i = tid; i < CC * 8; i += 256) {
    int row = i >> 3, q = i & 7;
    ((uint4*)sWre)[row * 9 + q] = gWre[row * 8 + q];
    ((uint4*)sWim)[row * 9 + q] = gWim[row * 8 + q];
    ((uint4*)sWin)[row * 9 + q] = gWin[row * 8 + q];
  }
#endif
  __syncthreads();

  int wave = tid >> 5;
  int lane = tid & 31;
  int mrowl = lane & 15;     // A fragment: lanes 0..15 = rows 0..15
  int hlf   = lane >> 4;     // A fragment: lane half selects K sub-range
  int mt    = wave;          // 8 waves cover 8 M-subtiles of the 128-row block

  v8f zero = {0.f, 0.f, 0.f, 0.f, 0.f, 0.f, 0.f, 0.f};
  v8f accRe[4] = {zero, zero, zero, zero};
  v8f accIm[4] = {zero, zero, zero, zero};

  #pragma unroll
  for (int kk = 0; kk < CC; kk += 32) {
    // A fragments (16x32 bf16), ISA 7.12.2 layout:
    //   e<8:  K = hlf*8 + e ;  e>=8: K = 16 + hlf*8 + (e-8)
    v16bf aRe, aIm;
    #pragma unroll
    for (int e = 0; e < 16; ++e) {
      int k  = kk + ((e < 8) ? (hlf * 8 + e) : (16 + hlf * 8 + (e - 8)));
      int ai = (mt * 16 + mrowl) * XSTR + k;
      aRe[e] = bfbits(sXre[ai]);
      aIm[e] = bfbits(sXim[ai]);
    }

    #pragma unroll
    for (int nt = 0; nt < 4; ++nt) {
      v16bf bRe, bIm, bIn;
      #pragma unroll
      for (int e = 0; e < 16; ++e) {
        int wi = (kk + ((e < 8) ? (hlf * 8 + e) : (16 + hlf * 8 + (e - 8)))) * XSTR + nt * 16 + mrowl;
        bRe[e] = bfbits(sWre[wi]);
        bIm[e] = bfbits(sWim[wi]);
        bIn[e] = bfbits(sWin[wi]);
      }
      // Yre += Xre*Wre + Xim*(-Wim) ; Yim += Xre*Wim + Xim*Wre
      accRe[nt] = __builtin_amdgcn_wmma_f32_16x16x32_bf16(false, aRe, false, bRe, (short)0, accRe[nt], false, false);
      accRe[nt] = __builtin_amdgcn_wmma_f32_16x16x32_bf16(false, aIm, false, bIn, (short)0, accRe[nt], false, false);
      accIm[nt] = __builtin_amdgcn_wmma_f32_16x16x32_bf16(false, aRe, false, bIm, (short)0, accIm[nt], false, false);
      accIm[nt] = __builtin_amdgcn_wmma_f32_16x16x32_bf16(false, aIm, false, bRe, (short)0, accIm[nt], false, false);
    }
  }

  // C/D layout: VGPR v -> M = v + 8*(lane/16), N = lane%16
  int n = lane & 15;
  int mbase = mb * MBLK + mt * 16 + 8 * hlf;
  #pragma unroll
  for (int nt = 0; nt < 4; ++nt) {
    #pragma unroll
    for (int v = 0; v < 8; ++v) {
      size_t yi = ((size_t)f * MR + mbase + v) * OO + nt * 16 + n;
      Yre[yi] = accRe[nt][v];
      Yim[yi] = accIm[nt][v];
    }
  }
}

// ---------------- kernel 4: inverse 2D transform + crop + scatter -----------
__global__ __launch_bounds__(256) void dw_inv(
    const float* __restrict__ Yre, const float* __restrict__ Yim,
    float* __restrict__ out)
{
  __shared__ float sMrr[FB], sMri[FB], sMir[FB], sMii[FB];
  __shared__ float sTre[OO * FB];  // 25.6 KB
  __shared__ float sTim[OO * FB];  // 25.6 KB
  int tid = threadIdx.x;
  if (tid < FB) {
    float a, b2, c2, d;
    dyadic_mats(tid / NN, tid % NN, true, a, b2, c2, d);
    sMrr[tid] = a; sMri[tid] = b2; sMir[tid] = c2; sMii[tid] = d;
  }
  __syncthreads();

  int m  = blockIdx.x;
  int b  = m / LT;
  int l  = m - b * LT;
  int ty = l / TWT;
  int tx = l - ty * TWT;

  // pass 1: transform along last axis (k -> j), scale 1/N
  for (int t = tid; t < OO * NN; t += 256) {
    int o = t & (OO - 1);           // o fastest -> coalesced Y loads
    int r = t >> 6;
    float yre[NN], yim[NN];
    #pragma unroll
    for (int k = 0; k < NN; ++k) {
      size_t yi = ((size_t)(r * NN + k) * MR + m) * OO + o;
      yre[k] = Yre[yi];
      yim[k] = Yim[yi];
    }
    #pragma unroll
    for (int j = 0; j < NN; ++j) {
      float re = 0.f, im = 0.f;
      #pragma unroll
      for (int k = 0; k < NN; ++k) {
        re += sMrr[j * NN + k] * yre[k] + sMri[j * NN + k] * yim[k];
        im += sMir[j * NN + k] * yre[k] + sMii[j * NN + k] * yim[k];
      }
      sTre[o * FB + r * NN + j] = re * 0.1f;
      sTim[o * FB + r * NN + j] = im * 0.1f;
    }
  }
  __syncthreads();

  // pass 2: transform along rows (r -> i), real part only, crop center 8x8
  for (int t = tid; t < OO * MT; t += 256) {
    int o = t & (OO - 1);
    int i = 1 + (t >> 6);           // kept rows 1..8
    float res[MT];
    #pragma unroll
    for (int j = 1; j <= MT; ++j) { // kept cols 1..8
      float acc = 0.f;
      #pragma unroll
      for (int r = 0; r < NN; ++r) {
        acc += sMrr[i * NN + r] * sTre[o * FB + r * NN + j]
             + sMri[i * NN + r] * sTim[o * FB + r * NN + j];
      }
      res[j - 1] = acc * 0.1f;
    }
    float* orow = out + (((size_t)b * OO + o) * HWD + (ty * MT + i - 1)) * HWD + tx * MT;
    #pragma unroll
    for (int jj = 0; jj < MT; ++jj) orow[jj] = res[jj];
  }
}

// ---------------- launch -----------------------------------------------------
extern "C" void kernel_launch(void* const* d_in, const int* in_sizes, int n_in,
                              void* d_out, int out_size, void* d_ws, size_t ws_size,
                              hipStream_t stream) {
  (void)in_sizes; (void)n_in; (void)out_size; (void)ws_size;
  const float* x   = (const float*)d_in[0];
  const float* wre = (const float*)d_in[1];
  const float* wim = (const float*)d_in[2];
  float* out = (float*)d_out;

  // Workspace layout (bytes, all 16B aligned):
  //  Xre/Xim bf16: 2 * 80,281,600   W bf16 x3: 3 * 819,200
  //  Yre/Yim f32 : 2 * 160,563,200  total ~462 MiB
  char* ws = (char*)d_ws;
  const size_t nX = (size_t)FB * MR * CC;
  const size_t nW = (size_t)FB * CC * OO;
  const size_t nY = (size_t)FB * MR * OO;
  u16* Xre = (u16*)ws;
  u16* Xim = Xre + nX;
  u16* Wre = Xim + nX;
  u16* Wim = Wre + nW;
  u16* Win = Wim + nW;
  float* Yre = (float*)(Win + nW);
  float* Yim = Yre + nY;

  dw_prep_weights<<<(FB * CC * OO + 255) / 256, 256, 0, stream>>>(wre, wim, Wre, Wim, Win);
  dw_fwd<<<MR, 256, 0, stream>>>(x, Xre, Xim);
  dw_gemm<<<dim3(NMB, FB), 256, 0, stream>>>(Xre, Xim, Wre, Wim, Win, Yre, Yim);
  dw_inv<<<MR, 256, 0, stream>>>(Yre, Yim, out);
}
